// ImplicitDAM_5325759447207
// MI455X (gfx1250) — matrix operand order repaired and checked by reading, not compiled, not run on hardware
//
#include <hip/hip_runtime.h>
#include <hip/hip_bf16.h>

// Problem constants (match reference)
#define B     2048
#define DIN   512
#define DCOMB 1024
#define DK    64
#define DV    64
#define NB    65536
#define LAM   0.1f
#define CH    16      // softmax stat chunks per row
#define RS    8       // bin splits for r-kernel

typedef __attribute__((ext_vector_type(16))) __bf16 v16bf;
typedef __attribute__((ext_vector_type(8)))  __bf16 v8bf;
typedef __attribute__((ext_vector_type(8)))  float  v8f;

union V16U { v16bf v; v8bf h[2]; };

__device__ __forceinline__ v8f wmma_bf16(v16bf a, v16bf b, v8f c) {
  // emits v_wmma_f32_16x16x32_bf16
  return __builtin_amdgcn_wmma_f32_16x16x32_bf16(false, a, false, b, (short)0, c, false, false);
}

// ---- workspace layout (bytes) ----
#define WS_K2    0u         // B*64 bf16  (2*k, bf16)        262144
#define WS_MT    262144u    // 64*B bf16  (m transposed)     262144
#define WS_P2    524288u    // NB  f32    (|p|^2)            262144
#define WS_ST    786432u    // B*CH*2 f32 (partial max/sum)  262144
#define WS_RM    1048576u   // B f32 row max
#define WS_RZ    1056768u   // B f32 row 1/Z
#define WS_PBF   2097152u   // NB*64 bf16 prototypes         8388608
#define WS_BT    10485760u  // 64*NB bf16 bufferT            8388608
// total ~18 MB

// ------------------------------------------------------------------
// 1) projections: k2 = 2*(Wk@comb + bk) -> bf16 [B][64]; mT bf16 [64][B]
__global__ void k_proj(const float* __restrict__ x, const float* __restrict__ ctx,
                       const float* __restrict__ Wk, const float* __restrict__ bk,
                       const float* __restrict__ Wc, const float* __restrict__ bc,
                       __bf16* __restrict__ k2bf, __bf16* __restrict__ mtbf) {
  int row = blockIdx.x;          // B rows
  int t   = threadIdx.x;         // 64 threads = 64 output features
  const float* xr = x   + (size_t)row * DIN;
  const float* cr = ctx + (size_t)row * DIN;
  const float* wk = Wk + (size_t)t * DCOMB;
  const float* wc = Wc + (size_t)t * DCOMB;
  float ak = bk[t], am = bc[t];
  for (int d = 0; d < DIN; ++d) { float v = xr[d]; ak += v * wk[d];       am += v * wc[d]; }
  for (int d = 0; d < DIN; ++d) { float v = cr[d]; ak += v * wk[DIN + d]; am += v * wc[DIN + d]; }
  k2bf[(size_t)row * DK + t] = (__bf16)(2.0f * ak);
  mtbf[(size_t)t * B + row]  = (__bf16)am;
}

// 2) prototypes -> bf16, |p|^2, buffer -> bf16 transposed
__global__ void k_prep(const float* __restrict__ proto, const float* __restrict__ buf,
                       __bf16* __restrict__ pbf, __bf16* __restrict__ btbf,
                       float* __restrict__ p2) {
  int bin = blockIdx.x * blockDim.x + threadIdx.x;   // NB threads
  float s = 0.0f;
  #pragma unroll 8
  for (int j = 0; j < DK; ++j) {
    float v = proto[(size_t)bin * DK + j];
    s += v * v;
    pbf[(size_t)bin * DK + j]  = (__bf16)v;
    btbf[(size_t)j * NB + bin] = (__bf16)buf[(size_t)bin * DV + j];
  }
  p2[bin] = s;
}

// 3) zero the r output region (it is accumulated via atomics)
__global__ void k_zero(float* __restrict__ p, int n) {
  int i = blockIdx.x * blockDim.x + threadIdx.x;
  if (i < n) p[i] = 0.0f;
}

// ------------------------------------------------------------------
// 4) pass A: per-(16 rows, 4096 bins) online softmax stats via WMMA logits
__global__ void k_stats(const __bf16* __restrict__ k2bf, const __bf16* __restrict__ pbf,
                        const float* __restrict__ p2, float* __restrict__ stats) {
  int lane = threadIdx.x, lo = lane & 15, hi = lane >> 4;
  int rt = blockIdx.x, chunk = blockIdx.y;
  const int BPC = NB / CH;   // 4096 bins per chunk

  // A operand: k2 rows for this 16-row tile, K=64 in two 32-chunks
  const __bf16* krow = k2bf + (size_t)(rt * 16 + lo) * DK;
  V16U a0, a1;
  a0.h[0] = *(const v8bf*)(krow + 0  + hi * 16);
  a0.h[1] = *(const v8bf*)(krow + 8  + hi * 16);
  a1.h[0] = *(const v8bf*)(krow + 32 + hi * 16);
  a1.h[1] = *(const v8bf*)(krow + 40 + hi * 16);

  float rmax[8], rsum[8];
  #pragma unroll
  for (int i = 0; i < 8; ++i) { rmax[i] = -3.0e38f; rsum[i] = 0.0f; }

  for (int nb = 0; nb < BPC / 16; ++nb) {
    int nbase = chunk * BPC + nb * 16;
    const __bf16* prow = pbf + (size_t)(nbase + lo) * DK;
    v16bf b0 = *(const v16bf*)(prow + hi * 16);
    v16bf b1 = *(const v16bf*)(prow + 32 + hi * 16);
    v8f c = {0.f,0.f,0.f,0.f,0.f,0.f,0.f,0.f};
    c = wmma_bf16(a0.v, b0, c);
    c = wmma_bf16(a1.v, b1, c);
    float p2l = p2[nbase + lo];
    #pragma unroll
    for (int i = 0; i < 8; ++i) {
      float l  = c[i] - p2l;                 // logit (||k||^2 cancels in softmax)
      float nm = fmaxf(rmax[i], l);
      rsum[i]  = rsum[i] * __expf(rmax[i] - nm) + __expf(l - nm);
      rmax[i]  = nm;
    }
  }
  // butterfly reduce (max,sum) within each 16-lane half
  #pragma unroll
  for (int m = 1; m < 16; m <<= 1) {
    #pragma unroll
    for (int i = 0; i < 8; ++i) {
      float om = __shfl_xor(rmax[i], m, 16);
      float os = __shfl_xor(rsum[i], m, 16);
      float nm = fmaxf(rmax[i], om);
      rsum[i]  = rsum[i] * __expf(rmax[i] - nm) + os * __expf(om - nm);
      rmax[i]  = nm;
    }
  }
  if (lo == 0) {
    #pragma unroll
    for (int i = 0; i < 8; ++i) {
      int row = rt * 16 + i + 8 * hi;
      stats[((size_t)row * CH + chunk) * 2 + 0] = rmax[i];
      stats[((size_t)row * CH + chunk) * 2 + 1] = rsum[i];
    }
  }
}

// 5) reduce CH partial (max,sum) pairs -> rowM, rowInvZ
__global__ void k_stats2(const float* __restrict__ stats,
                         float* __restrict__ rowM, float* __restrict__ rowZ) {
  int row = blockIdx.x * blockDim.x + threadIdx.x;
  if (row >= B) return;
  float m = -3.0e38f, s = 0.0f;
  for (int c = 0; c < CH; ++c) {
    float om = stats[((size_t)row * CH + c) * 2 + 0];
    float os = stats[((size_t)row * CH + c) * 2 + 1];
    float nm = fmaxf(m, om);
    s = s * __expf(m - nm) + os * __expf(om - nm);
    m = nm;
  }
  rowM[row] = m;
  rowZ[row] = 1.0f / s;
}

// ------------------------------------------------------------------
// 6) r = a @ buffer : recompute logits tile, a = softmax, WMMA against bufT
__global__ void k_r(const __bf16* __restrict__ k2bf, const __bf16* __restrict__ pbf,
                    const float* __restrict__ p2, const __bf16* __restrict__ btbf,
                    const float* __restrict__ rowM, const float* __restrict__ rowZ,
                    float* __restrict__ rout) {
  __shared__ __align__(32) unsigned short ldsraw[16 * 32];
  __bf16* lds = (__bf16*)ldsraw;
  int lane = threadIdx.x, lo = lane & 15, hi = lane >> 4;
  int rt = blockIdx.x, split = blockIdx.y;
  const int BPS = NB / RS;   // 8192 bins per split

  const __bf16* krow = k2bf + (size_t)(rt * 16 + lo) * DK;
  V16U a0, a1;
  a0.h[0] = *(const v8bf*)(krow + 0  + hi * 16);
  a0.h[1] = *(const v8bf*)(krow + 8  + hi * 16);
  a1.h[0] = *(const v8bf*)(krow + 32 + hi * 16);
  a1.h[1] = *(const v8bf*)(krow + 40 + hi * 16);

  float Mv[8], Zv[8];
  #pragma unroll
  for (int i = 0; i < 8; ++i) {
    int row = rt * 16 + i + 8 * hi;
    Mv[i] = rowM[row]; Zv[i] = rowZ[row];
  }
  v8f racc[4];
  #pragma unroll
  for (int t = 0; t < 4; ++t) racc[t] = (v8f){0.f,0.f,0.f,0.f,0.f,0.f,0.f,0.f};

  for (int kb = 0; kb < BPS / 32; ++kb) {
    int binb = split * BPS + kb * 32;
    #pragma unroll
    for (int sub = 0; sub < 2; ++sub) {
      int nb = binb + sub * 16;
      const __bf16* prow = pbf + (size_t)(nb + lo) * DK;
      v16bf b0 = *(const v16bf*)(prow + hi * 16);
      v16bf b1 = *(const v16bf*)(prow + 32 + hi * 16);
      v8f c = {0.f,0.f,0.f,0.f,0.f,0.f,0.f,0.f};
      c = wmma_bf16(a0.v, b0, c);
      c = wmma_bf16(a1.v, b1, c);
      float p2l = p2[nb + lo];
      #pragma unroll
      for (int i = 0; i < 8; ++i) {
        float av = __expf(c[i] - p2l - Mv[i]) * Zv[i];
        lds[(i + 8 * hi) * 32 + sub * 16 + lo] = (__bf16)av;   // a tile, [batch][bin]
      }
    }
    asm volatile("s_wait_dscnt 0" ::: "memory");   // wave-local LDS staging
    V16U aa;   // a in A-layout (K = 32 bins)
    aa.h[0] = *(const v8bf*)(lds + lo * 32 + hi * 16);
    aa.h[1] = *(const v8bf*)(lds + lo * 32 + 8 + hi * 16);
    #pragma unroll
    for (int dvt = 0; dvt < 4; ++dvt) {
      const __bf16* brow = btbf + (size_t)(dvt * 16 + lo) * NB + binb + hi * 16;
      v16bf bb = *(const v16bf*)brow;
      racc[dvt] = wmma_bf16(aa.v, bb, racc[dvt]);
    }
  }
  #pragma unroll
  for (int dvt = 0; dvt < 4; ++dvt)
    #pragma unroll
    for (int i = 0; i < 8; ++i)
      atomicAdd(&rout[(size_t)(rt * 16 + i + 8 * hi) * DV + dvt * 16 + lo], racc[dvt][i]);
}

// ------------------------------------------------------------------
// 7) buffer update: per 16-bin tile, accumulate a^T@m and a_sum over all batch
__global__ void k_u(const __bf16* __restrict__ k2bf, const __bf16* __restrict__ pbf,
                    const float* __restrict__ p2, const __bf16* __restrict__ mtbf,
                    const float* __restrict__ rowM, const float* __restrict__ rowZ,
                    const float* __restrict__ buf, float* __restrict__ nbout) {
  __shared__ __align__(32) unsigned short ldsraw[16 * 32];
  __bf16* lds = (__bf16*)ldsraw;
  int lane = threadIdx.x, lo = lane & 15, hi = lane >> 4;
  int binb = blockIdx.x * 16;

  // A operand: prototype rows (bins x K=64)
  const __bf16* prow = pbf + (size_t)(binb + lo) * DK;
  V16U a0, a1;
  a0.h[0] = *(const v8bf*)(prow + 0  + hi * 16);
  a0.h[1] = *(const v8bf*)(prow + 8  + hi * 16);
  a1.h[0] = *(const v8bf*)(prow + 32 + hi * 16);
  a1.h[1] = *(const v8bf*)(prow + 40 + hi * 16);

  float pv[8];
  #pragma unroll
  for (int i = 0; i < 8; ++i) pv[i] = p2[binb + i + 8 * hi];

  v8f nacc[4];
  #pragma unroll
  for (int t = 0; t < 4; ++t) nacc[t] = (v8f){0.f,0.f,0.f,0.f,0.f,0.f,0.f,0.f};
  float asum[8];
  #pragma unroll
  for (int i = 0; i < 8; ++i) asum[i] = 0.0f;

  for (int bb = 0; bb < B / 32; ++bb) {
    int bbase = bb * 32;
    #pragma unroll
    for (int sub = 0; sub < 2; ++sub) {
      int bc = bbase + sub * 16;
      const __bf16* krow = k2bf + (size_t)(bc + lo) * DK;   // B operand: k2 rows contiguous
      v16bf b0 = *(const v16bf*)(krow + hi * 16);
      v16bf b1 = *(const v16bf*)(krow + 32 + hi * 16);
      v8f c = {0.f,0.f,0.f,0.f,0.f,0.f,0.f,0.f};
      c = wmma_bf16(a0.v, b0, c);     // logits^T tile [bin x batch]
      c = wmma_bf16(a1.v, b1, c);
      float Mb = rowM[bc + lo], Zb = rowZ[bc + lo];
      #pragma unroll
      for (int i = 0; i < 8; ++i) {
        float av = __expf(c[i] - pv[i] - Mb) * Zb;
        asum[i] += av;
        lds[(i + 8 * hi) * 32 + sub * 16 + lo] = (__bf16)av;  // a^T tile [bin][batch]
      }
    }
    asm volatile("s_wait_dscnt 0" ::: "memory");
    V16U aa;   // a^T in A-layout (K = 32 batch)
    aa.h[0] = *(const v8bf*)(lds + lo * 32 + hi * 16);
    aa.h[1] = *(const v8bf*)(lds + lo * 32 + 8 + hi * 16);
    #pragma unroll
    for (int dvt = 0; dvt < 4; ++dvt) {
      const __bf16* mrow = mtbf + (size_t)(dvt * 16 + lo) * B + bbase + hi * 16;
      v16bf mb = *(const v16bf*)mrow;
      nacc[dvt] = wmma_bf16(aa.v, mb, nacc[dvt]);
    }
  }
  // a_sum: reduce over the 16 batch-columns held by this lane-half
  #pragma unroll
  for (int m = 1; m < 16; m <<= 1)
    #pragma unroll
    for (int i = 0; i < 8; ++i)
      asum[i] += __shfl_xor(asum[i], m, 16);

  const float cc = LAM / (float)B;
  #pragma unroll
  for (int dvt = 0; dvt < 4; ++dvt)
    #pragma unroll
    for (int i = 0; i < 8; ++i) {
      int bin = binb + i + 8 * hi;
      int dv  = dvt * 16 + lo;
      size_t idx = (size_t)bin * DV + dv;
      float bv = buf[idx];
      nbout[idx] = bv + cc * (nacc[dvt][i] - asum[i] * bv);
    }
}

// ------------------------------------------------------------------
extern "C" void kernel_launch(void* const* d_in, const int* in_sizes, int n_in,
                              void* d_out, int out_size, void* d_ws, size_t ws_size,
                              hipStream_t stream) {
  const float* x     = (const float*)d_in[0];
  const float* ctx   = (const float*)d_in[1];
  const float* Wk    = (const float*)d_in[2];
  const float* bk    = (const float*)d_in[3];
  const float* Wc    = (const float*)d_in[4];
  const float* bc    = (const float*)d_in[5];
  const float* proto = (const float*)d_in[6];
  const float* buf   = (const float*)d_in[7];

  float* rout  = (float*)d_out;            // [B, DV]
  float* nbout = rout + (size_t)B * DV;    // [NB, DV]

  char* ws = (char*)d_ws;
  __bf16* k2bf = (__bf16*)(ws + WS_K2);
  __bf16* mtbf = (__bf16*)(ws + WS_MT);
  float*  p2   = (float*)(ws + WS_P2);
  float*  st   = (float*)(ws + WS_ST);
  float*  rowM = (float*)(ws + WS_RM);
  float*  rowZ = (float*)(ws + WS_RZ);
  __bf16* pbf  = (__bf16*)(ws + WS_PBF);
  __bf16* btbf = (__bf16*)(ws + WS_BT);

  k_proj <<<dim3(B), dim3(64), 0, stream>>>(x, ctx, Wk, bk, Wc, bc, k2bf, mtbf);
  k_prep <<<dim3(NB / 256), dim3(256), 0, stream>>>(proto, buf, pbf, btbf, p2);
  k_zero <<<dim3((B * DV + 255) / 256), dim3(256), 0, stream>>>(rout, B * DV);
  k_stats<<<dim3(B / 16, CH), dim3(32), 0, stream>>>(k2bf, pbf, p2, st);
  k_stats2<<<dim3((B + 255) / 256), dim3(256), 0, stream>>>(st, rowM, rowZ);
  k_r    <<<dim3(B / 16, RS), dim3(32), 0, stream>>>(k2bf, pbf, p2, btbf, rowM, rowZ, rout);
  k_u    <<<dim3(NB / 16), dim3(32), 0, stream>>>(k2bf, pbf, p2, mtbf, rowM, rowZ, buf, nbout);
}
